// MaxLuk2LAFRNN_Cifar10_52802327937295
// MI455X (gfx1250) — compile-verified, matrix operand order checked by reading, not run on hardware
//
#include <hip/hip_runtime.h>
#include <math.h>

typedef _Float16 v16h __attribute__((ext_vector_type(16)));
typedef _Float16 v8h  __attribute__((ext_vector_type(8)));
typedef float    v8f  __attribute__((ext_vector_type(8)));

// Async global->LDS copy of one 16B chunk per lane (CDNA5 GLOBAL_LOAD_ASYNC_TO_LDS_B128).
// ldsOff: byte offset within wave/WG LDS allocation; gaddr: 64-bit global address.
__device__ __forceinline__ void async_copy_b128(unsigned ldsOff, const void* gaddr) {
    asm volatile("global_load_async_to_lds_b128 %0, %1, off"
                 :: "v"(ldsOff), "v"(gaddr) : "memory");
}
__device__ __forceinline__ void wait_async_all() {
    asm volatile("s_wait_asynccnt 0x0" ::: "memory");
}
// Generic (flat) pointer to LDS aperture: low 32 bits are the LDS byte offset.
__device__ __forceinline__ unsigned lds_off(const void* p) {
    return (unsigned)(unsigned long long)p;
}

// ---------------------------------------------------------------------------
// Stage 1: fuzzy conv1 (Lukasiewicz / max-plus).  x:[256,3,32,32] w1:[8,3,3,3]
// -> x1:[256,8,30,30] f32.  One block per image; image staged via async LDS DMA.
// ---------------------------------------------------------------------------
__global__ void fuzzy_conv1(const float* __restrict__ x,
                            const float* __restrict__ w1,
                            float* __restrict__ x1) {
    __shared__ float xs[3 * 32 * 32];   // 12288 B = exactly 3 sweeps of 256x16B
    __shared__ float ws[8 * 27];
    const int b   = blockIdx.x;
    const int tid = threadIdx.x;

    const char* src = (const char*)(x + b * 3072);
    const unsigned xs0 = lds_off(xs);
    #pragma unroll
    for (int s = 0; s < 3; ++s) {
        const unsigned off = tid * 16 + s * 4096;
        async_copy_b128(xs0 + off, src + off);
    }
    for (int i = tid; i < 8 * 27; i += 256) ws[i] = w1[i];
    wait_async_all();
    __syncthreads();

    for (int p = tid; p < 8 * 30 * 30; p += 256) {
        const int o   = p / 900;
        const int rem = p - o * 900;
        const int i   = rem / 30;
        const int j   = rem - i * 30;
        float m = -1e30f;
        #pragma unroll
        for (int c = 0; c < 3; ++c) {
            const float* xb = &xs[c * 1024 + i * 32 + j];
            const float* wb = &ws[o * 27 + c * 9];
            #pragma unroll
            for (int kh = 0; kh < 3; ++kh)
                #pragma unroll
                for (int kw = 0; kw < 3; ++kw)
                    m = fmaxf(m, xb[kh * 32 + kw] + wb[kh * 3 + kw]);
        }
        x1[b * 7200 + p] = fmaxf(m - 1.0f, 0.0f);
    }
}

// ---------------------------------------------------------------------------
// Stage 2: fuzzy conv2.  x1:[256,8,30,30] w2:[16,8,3,3] -> x2:[256,16,28,28]
// written directly as f16 (GEMM operand A).  Input tile staged via async DMA.
// ---------------------------------------------------------------------------
__global__ void fuzzy_conv2(const float* __restrict__ x1,
                            const float* __restrict__ w2,
                            _Float16* __restrict__ x2h) {
    __shared__ float xs[8 * 30 * 30];   // 28800 B = 7 full sweeps + 128 B tail
    __shared__ float ws[16 * 72];
    const int b   = blockIdx.x;
    const int tid = threadIdx.x;

    const char* src = (const char*)(x1 + b * 7200);
    const unsigned xs0 = lds_off(xs);
    for (unsigned off = tid * 16; off < 28800u; off += 4096u)
        async_copy_b128(xs0 + off, src + off);
    for (int i = tid; i < 16 * 72; i += 256) ws[i] = w2[i];
    wait_async_all();
    __syncthreads();

    for (int p = tid; p < 16 * 28 * 28; p += 256) {
        const int o   = p / 784;
        const int rem = p - o * 784;
        const int i   = rem / 28;
        const int j   = rem - i * 28;
        float m = -1e30f;
        #pragma unroll
        for (int c = 0; c < 8; ++c) {
            const float* xb = &xs[c * 900 + i * 30 + j];
            const float* wb = &ws[o * 72 + c * 9];
            #pragma unroll
            for (int kh = 0; kh < 3; ++kh)
                #pragma unroll
                for (int kw = 0; kw < 3; ++kw)
                    m = fmaxf(m, xb[kh * 30 + kw] + wb[kh * 3 + kw]);
        }
        x2h[b * 12544 + p] = (_Float16)fmaxf(m - 1.0f, 0.0f);
    }
}

// ---------------------------------------------------------------------------
// Stage 3: wd [10,12544] f32 -> f16 padded to [16,12544] (rows >=10 zero).
// ---------------------------------------------------------------------------
__global__ void cvt_wd_f16(const float* __restrict__ wd,
                           _Float16* __restrict__ wdh) {
    const int idx = blockIdx.x * 256 + threadIdx.x;
    if (idx >= 16 * 12544) return;
    const int n = idx / 12544;
    wdh[idx] = (n < 10) ? (_Float16)wd[idx] : (_Float16)0.0f;
}

// ---------------------------------------------------------------------------
// Stage 4: logits = x2h @ wdh^T via v_wmma_f32_16x16x32_f16 with split-K,
// then per-row log_softmax.  16 blocks x 256 threads (8 waves). Wave w owns
// K-chunk [w*1568, w*1568+1568) -> 49 WMMAs; the 8 partial 16x16 C tiles are
// reduced in LDS.  A-fragment layout (ISA 7.12.2, 16-bit A 16x32):
//   lane L: row M = L%16; halves 0..7  <- K = k0 + (L/16)*8 .. +7
//           halves 8..15 <- K = k0 + 16 + (L/16)*8 .. +7   (two b128 loads)
// ---------------------------------------------------------------------------
__global__ void dense_logsoftmax(const _Float16* __restrict__ x2h,
                                 const _Float16* __restrict__ wdh,
                                 float* __restrict__ out) {
    __shared__ float partial[8][16][16];
    __shared__ float tile[16][17];
    const int tid  = threadIdx.x;
    const int wv   = tid >> 5;             // wave 0..7 (K-split)
    const int lane = tid & 31;             // all 32 lanes active for WMMA
    const int rt   = lane & 15;            // A row-in-tile / B column
    const int hs8  = (lane >> 4) * 8;      // half-select K offset

    const _Float16* Arow = x2h + (blockIdx.x * 16 + rt) * 12544 + wv * 1568;
    const _Float16* Brow = wdh + rt * 12544 + wv * 1568;

    v8f acc = {};
    #pragma unroll 7
    for (int k0 = 0; k0 < 1568; k0 += 32) {
        v8h alo = *(const v8h*)(Arow + k0 + hs8);
        v8h ahi = *(const v8h*)(Arow + k0 + 16 + hs8);
        v8h blo = *(const v8h*)(Brow + k0 + hs8);
        v8h bhi = *(const v8h*)(Brow + k0 + 16 + hs8);
        v16h a  = __builtin_shufflevector(alo, ahi, 0,1,2,3,4,5,6,7,8,9,10,11,12,13,14,15);
        v16h bm = __builtin_shufflevector(blo, bhi, 0,1,2,3,4,5,6,7,8,9,10,11,12,13,14,15);
        // (neg_a, A, neg_b, B, c_mod, C, reuse_a, reuse_b)
        acc = __builtin_amdgcn_wmma_f32_16x16x32_f16(
            false, a, false, bm, (short)0, acc, false, false);
    }

    // C layout: VGPR r holds M = r + (lane/16)*8, N = lane%16
    #pragma unroll
    for (int r = 0; r < 8; ++r)
        partial[wv][r + hs8][rt] = acc[r];
    __syncthreads();

    {   // split-K reduction: one thread per (m,n) element
        const int m = tid >> 4, n = tid & 15;
        float s = 0.0f;
        #pragma unroll
        for (int w = 0; w < 8; ++w) s += partial[w][m][n];
        tile[m][n] = s;
    }
    __syncthreads();

    if (tid < 16) {
        float m = -1e30f;
        #pragma unroll
        for (int n = 0; n < 10; ++n) m = fmaxf(m, tile[tid][n]);
        float s = 0.0f;
        #pragma unroll
        for (int n = 0; n < 10; ++n) s += __expf(tile[tid][n] - m);
        const float ls = __logf(s);
        const int row = blockIdx.x * 16 + tid;
        #pragma unroll
        for (int n = 0; n < 10; ++n)
            out[row * 10 + n] = tile[tid][n] - m - ls;
    }
}

// ---------------------------------------------------------------------------
// Workspace layout (bytes):
//   [0,        7372800)  x1  f32 [256,8,30,30]
//   [7372800, 13795328)  x2h f16 [256,12544]
//   [13795328,14196736)  wdh f16 [16,12544]
// ---------------------------------------------------------------------------
extern "C" void kernel_launch(void* const* d_in, const int* in_sizes, int n_in,
                              void* d_out, int out_size, void* d_ws, size_t ws_size,
                              hipStream_t stream) {
    const float* x  = (const float*)d_in[0];   // [256,3,32,32]
    const float* w1 = (const float*)d_in[1];   // [8,3,3,3]
    const float* w2 = (const float*)d_in[2];   // [16,8,3,3]
    const float* wd = (const float*)d_in[3];   // [10,12544]
    float* out = (float*)d_out;                // [256,10]

    char* ws = (char*)d_ws;
    float*    x1  = (float*)(ws);
    _Float16* x2h = (_Float16*)(ws + 7372800);
    _Float16* wdh = (_Float16*)(ws + 13795328);

    fuzzy_conv1<<<dim3(256), dim3(256), 0, stream>>>(x, w1, x1);
    fuzzy_conv2<<<dim3(256), dim3(256), 0, stream>>>(x1, w2, x2h);
    cvt_wd_f16<<<dim3((16 * 12544 + 255) / 256), dim3(256), 0, stream>>>(wd, wdh);
    dense_logsoftmax<<<dim3(16), dim3(256), 0, stream>>>(x2h, wdh, out);
}
